// DictionaryLearningBottleneck_43456479101593
// MI455X (gfx1250) — matrix operand order repaired and checked by reading, not compiled, not run on hardware
//
#include <hip/hip_runtime.h>
#include <stdint.h>

typedef __bf16 v16bf __attribute__((ext_vector_type(16)));
typedef float  v8f   __attribute__((ext_vector_type(8)));

#define EMBED   64
#define KATOMS  512
#define NSIG    32768
#define NT      128
#define SPARS   5

#define ZQ_OFF   0
#define LOSS_OFF 2097152
#define COEF_OFF 2097153

// ---------------------------------------------------------------------------
// Zero the loss slot + coefficients region of d_out (harness poisons buffers).
// ---------------------------------------------------------------------------
__global__ void fill_zero_kernel(float* __restrict__ out) {
  int idx = blockIdx.x * 256 + threadIdx.x;
  if (idx < 16777217) out[LOSS_OFF + idx] = 0.0f;
}

// ---------------------------------------------------------------------------
// Build Dt (512x64 f32 row-major) and the bf16 A-fragment pack of Dt laid out
// in contiguous k order (k = 16g + i, same as the B loads) (lane-contiguous 32B fragments).
//   Apack[((tm*2+kt)*32 + lane)*16 + i]  with
//   M = lane&15, g = lane>>4, v = i>>1, pos = i&1
//   Klocal = 16g + i
// ---------------------------------------------------------------------------
__global__ void prep_pack_kernel(const float* __restrict__ D,
                                 float* __restrict__ Dt,
                                 __bf16* __restrict__ Apack) {
  int p = blockIdx.x * 256 + threadIdx.x;   // 0..32767
  // Dt: p = a*64 + e
  {
    int a = p >> 6, e = p & 63;
    Dt[p] = D[e * KATOMS + a];
  }
  // Apack
  {
    int i = p & 15, lane = (p >> 4) & 31, kt = (p >> 9) & 1, tm = p >> 10;
    int Ml = lane & 15, g = lane >> 4, v = i >> 1, pos = i & 1;
    int Kl = 16 * g + i; (void)v; (void)pos;
    int a = tm * 16 + Ml;
    int e = kt * 32 + Kl;
    Apack[p] = (__bf16)D[e * KATOMS + a];
  }
}

// ---------------------------------------------------------------------------
// DtD = Dt @ D + eps*I  (512x512 f32; tiny, stays L2-resident)
// ---------------------------------------------------------------------------
__global__ void prep_dtd_kernel(const float* __restrict__ Dt,
                                float* __restrict__ DtD) {
  int p = blockIdx.x * 256 + threadIdx.x;   // 0..262143
  int i = p >> 9, j = p & 511;
  float s = 0.0f;
  #pragma unroll 8
  for (int e = 0; e < EMBED; ++e) s += Dt[i * EMBED + e] * Dt[j * EMBED + e];
  DtD[p] = s + ((i == j) ? 1e-10f : 0.0f);
}

// ---------------------------------------------------------------------------
// Fused batch-OMP: 128 signals/block, 4 waves. corr via bf16 WMMA with fused
// masked argmax; per-signal padded 5x5 solve; residual update; final outputs.
// LDS ~126 KB/block -> two blocks (8 waves) co-resident per 320 KB WGP.
// ---------------------------------------------------------------------------
__global__ __launch_bounds__(128) void omp_main_kernel(
    const float* __restrict__ z, const float* __restrict__ DtD,
    const float* __restrict__ Dt, const uint32_t* __restrict__ ApackG,
    float* __restrict__ out, float* __restrict__ partials)
{
  __shared__ float                  s_sig[NT][EMBED + 1];   // 33.3 KB
  __shared__ __align__(32) __bf16   s_res[NT][EMBED];       // 16 KB
  __shared__ __align__(32) uint32_t s_ap[16384];            // 64 KB A-pack
  __shared__ uint32_t               s_excl[NT][16];         // 8 KB exclude bits
  __shared__ int                    s_sel[NT][SPARS];
  __shared__ float                  s_mv[NT][2];
  __shared__ int                    s_mi[NT][2];
  __shared__ float                  s_red[NT];

  const int t    = threadIdx.x;
  const int wid  = t >> 5;
  const int lane = t & 31;
  const int n    = blockIdx.x * NT + t;
  const int bb   = n >> 10;
  const int low  = n & 1023;
  const float* zcol = z + bb * 65536 + low;

  // Stage the 64 KB A-pack straight into LDS with CDNA5 async copies
  // (GLOBAL_LOAD_ASYNC_TO_LDS_B128, GVS mode, tracked by ASYNCcnt) while the
  // signal-column loads below proceed through the normal VMEM path.
  {
    uint32_t lds_base = (uint32_t)(uintptr_t)(&s_ap[0]);
    for (int it = t; it < 4096; it += NT) {          // 4096 x 16B = 64 KB
      uint32_t off = (uint32_t)it * 16u;
      asm volatile("global_load_async_to_lds_b128 %0, %1, %2"
                   :: "v"(lds_base + off), "v"(off), "s"(ApackG)
                   : "memory");
    }
  }

  // load signal column (stride-1024 f32, lane-coalesced) + bf16 residual init
  #pragma unroll 4
  for (int e = 0; e < EMBED; ++e) {
    float v = zcol[e * 1024];
    s_sig[t][e] = v;
    s_res[t][e] = (__bf16)v;
  }
  #pragma unroll
  for (int w = 0; w < 16; ++w) s_excl[t][w] = 0u;
  #pragma unroll
  for (int j = 0; j < SPARS; ++j) s_sel[t][j] = 0;

  asm volatile("s_wait_asynccnt 0x0" ::: "memory");
  __syncthreads();

  float c[SPARS];
  #pragma unroll
  for (int j = 0; j < SPARS; ++j) c[j] = 0.0f;
  float rec = 0.0f, l1 = 0.0f;

  for (int k = 0; k < SPARS; ++k) {
    // ---------------- WMMA corr + fused masked argmax ----------------
    const int col = lane & 15, g = lane >> 4;
    for (int sth = 0; sth < 2; ++sth) {
      const int st  = wid * 2 + sth;         // signal tile (16 signals)
      const int row = st * 16 + col;         // this lane's signal
      v16bf b0 = *(const v16bf*)&s_res[row][g * 16];
      v16bf b1 = *(const v16bf*)&s_res[row][32 + g * 16];
      float bestv = -3.4e38f;
      int   besti = 0;
      for (int tm = 0; tm < 32; ++tm) {      // 32 atom tiles of 16
        v16bf a0 = *(const v16bf*)&s_ap[(tm * 2 + 0) * 256 + lane * 8];
        v16bf a1 = *(const v16bf*)&s_ap[(tm * 2 + 1) * 256 + lane * 8];
        v8f acc = {};
        acc = __builtin_amdgcn_wmma_f32_16x16x32_bf16(false, a0, false, b0,
                                                      (short)0, acc, false, false);
        acc = __builtin_amdgcn_wmma_f32_16x16x32_bf16(false, a1, false, b1,
                                                      (short)0, acc, false, false);
        uint32_t w = s_excl[row][tm >> 1];
        uint32_t byteM = (w >> (((tm & 1) << 4) + (g << 3))) & 0xFFu;
        #pragma unroll
        for (int r = 0; r < 8; ++r) {        // lane covers atoms tm*16+8g+r
          float v = acc[r];
          if ((((byteM >> r) & 1u) == 0u) && v > bestv) {
            bestv = v;
            besti = tm * 16 + g * 8 + r;
          }
        }
      }
      s_mv[row][g] = bestv;
      s_mi[row][g] = besti;
    }
    __syncthreads();

    // ---------------- per-signal scalar phase ----------------
    {
      float v0 = s_mv[t][0], v1 = s_mv[t][1];
      int a = (v1 > v0) ? s_mi[t][1] : s_mi[t][0];
      s_sel[t][k] = a;
      s_excl[t][a >> 5] |= (1u << (a & 31));
      __builtin_prefetch(&Dt[a * EMBED], 0, 1);

      const int kk = k + 1;
      int sel[SPARS];
      #pragma unroll
      for (int j = 0; j < SPARS; ++j) sel[j] = s_sel[t][j];

      // Gram padded to 5x5 with identity -> fully unrolled solve, all regs
      float G[SPARS][SPARS], yv[SPARS];
      #pragma unroll
      for (int i = 0; i < SPARS; ++i) {
        #pragma unroll
        for (int j = 0; j < SPARS; ++j) {
          float gij = (i == j) ? 1.0f : 0.0f;
          if (i < kk && j < kk) gij = DtD[sel[i] * KATOMS + sel[j]];
          G[i][j] = gij;
        }
      }
      // y_i = d_i . residual_{k-1} = d_i.s - sum_j cprev_j * DtD[sel_i][sel_j]
      #pragma unroll
      for (int i = 0; i < SPARS; ++i) {
        float y = 0.0f;
        if (i < kk) {
          float ds = 0.0f;
          #pragma unroll 8
          for (int e = 0; e < EMBED; ++e) ds += Dt[sel[i] * EMBED + e] * s_sig[t][e];
          y = ds;
          #pragma unroll
          for (int j = 0; j < SPARS; ++j) y -= c[j] * G[i][j];
        }
        yv[i] = y;
      }
      // Gaussian elimination (no pivot; SPD + identity padding), unrolled
      #pragma unroll
      for (int p = 0; p < SPARS; ++p) {
        float inv = 1.0f / G[p][p];
        #pragma unroll
        for (int j2 = p + 1; j2 < SPARS; ++j2) G[p][j2] *= inv;
        yv[p] *= inv;
        #pragma unroll
        for (int i2 = p + 1; i2 < SPARS; ++i2) {
          float f = G[i2][p];
          #pragma unroll
          for (int j2 = p + 1; j2 < SPARS; ++j2) G[i2][j2] -= f * G[p][j2];
          yv[i2] -= f * yv[p];
        }
      }
      #pragma unroll
      for (int p = SPARS - 1; p >= 0; --p) {
        float sacc = yv[p];
        #pragma unroll
        for (int j2 = p + 1; j2 < SPARS; ++j2) sacc -= G[p][j2] * c[j2];
        c[p] = sacc;
      }

      // residual update (+ final outputs on last iteration)
      const bool last = (k == SPARS - 1);
      for (int e = 0; e < EMBED; ++e) {
        float u = 0.0f;
        #pragma unroll
        for (int j = 0; j < SPARS; ++j) u += c[j] * Dt[sel[j] * EMBED + e];
        float s0 = s_sig[t][e];
        float r = s0 - u;
        s_res[t][e] = (__bf16)r;
        if (last) {
          rec += r * r;
          float d = u - s0;
          d = fminf(fmaxf(d, -1.0f), 1.0f);      // straight-through clip
          out[ZQ_OFF + bb * 65536 + e * 1024 + low] = s0 + d;
        }
      }
      if (last) {
        #pragma unroll
        for (int j = 0; j < SPARS; ++j) {
          float cc = fminf(fmaxf(c[j], -1.0e7f), 1.0e7f);
          out[COEF_OFF + sel[j] * NSIG + n] = cc;
          l1 += fabsf(cc);
        }
      }
    }
    __syncthreads();
  }

  // deterministic block reductions for loss partials
  s_red[t] = rec; __syncthreads();
  if (t == 0) {
    float a = 0.0f;
    for (int i = 0; i < NT; ++i) a += s_red[i];
    partials[blockIdx.x] = a;
  }
  __syncthreads();
  s_red[t] = l1; __syncthreads();
  if (t == 0) {
    float a = 0.0f;
    for (int i = 0; i < NT; ++i) a += s_red[i];
    partials[256 + blockIdx.x] = a;
  }
}

// ---------------------------------------------------------------------------
// loss = 1.375 * mean(r^2)  + 0.001 * mean(|coeff|)
//  (rec_loss == commit_loss numerically; effective_cost = 0.25*(1+0.5))
// ---------------------------------------------------------------------------
__global__ void finalize_kernel(const float* __restrict__ partials,
                                float* __restrict__ out) {
  if (threadIdx.x == 0 && blockIdx.x == 0) {
    float recS = 0.0f, l1S = 0.0f;
    for (int i = 0; i < 256; ++i) recS += partials[i];
    for (int i = 0; i < 256; ++i) l1S += partials[256 + i];
    float rec_loss = recS / (float)(EMBED * NSIG);
    float coeff_loss = l1S / ((float)KATOMS * (float)NSIG);
    out[LOSS_OFF] = (1.0f + 0.375f) * rec_loss + 0.001f * coeff_loss;
  }
}

// ---------------------------------------------------------------------------
extern "C" void kernel_launch(void* const* d_in, const int* in_sizes, int n_in,
                              void* d_out, int out_size, void* d_ws, size_t ws_size,
                              hipStream_t stream) {
  (void)in_sizes; (void)n_in; (void)out_size; (void)ws_size;
  const float* z = (const float*)d_in[0];   // (32,64,32,32) f32
  const float* D = (const float*)d_in[1];   // (64,512) f32
  float* out = (float*)d_out;

  float* wsf      = (float*)d_ws;
  float* DtD      = wsf;                               // 262144 f32
  float* Dt       = wsf + 262144;                      // 32768 f32
  __bf16* Apack   = (__bf16*)(wsf + 262144 + 32768);   // 32768 bf16
  float* partials = wsf + 262144 + 32768 + 16384;      // 512 f32

  fill_zero_kernel<<<(16777217 + 255) / 256, 256, 0, stream>>>(out);
  prep_pack_kernel<<<32768 / 256, 256, 0, stream>>>(D, Dt, Apack);
  prep_dtd_kernel<<<262144 / 256, 256, 0, stream>>>(Dt, DtD);
  omp_main_kernel<<<NSIG / NT, NT, 0, stream>>>(z, DtD, Dt,
                                                (const uint32_t*)Apack, out, partials);
  finalize_kernel<<<1, 64, 0, stream>>>(partials, out);
}
